// SwitchFNN_30520037606033
// MI455X (gfx1250) — matrix operand order repaired and checked by reading, not compiled, not run on hardware
//
#include <hip/hip_runtime.h>
#include <hip/hip_bf16.h>
#include <math.h>

#define T_TOK   16384
#define DMODEL  512
#define DFF     2048
#define NEXP    8
#define M_TILE  64
#define FF_CHUNK 128

typedef __attribute__((ext_vector_type(16))) __bf16 v16bf;
typedef __attribute__((ext_vector_type(8)))  float  v8f;

// -------------------- 1) Router: logits -> softmax -> top-1 --------------------
__global__ __launch_bounds__(256) void router_kernel(
    const float* __restrict__ x, const float* __restrict__ sw_w,
    const float* __restrict__ sw_b, float* __restrict__ pmax_out,
    float* __restrict__ probs, int* __restrict__ routes)
{
    __shared__ float sw[NEXP][DMODEL];
    int tid = threadIdx.x;
    for (int i = tid; i < NEXP * DMODEL; i += 256)
        ((float*)sw)[i] = sw_w[i];
    __syncthreads();

    int lane = tid & 31, wid = tid >> 5;
    int t = blockIdx.x * 8 + wid;
    const float* xr = x + (size_t)t * DMODEL;

    float s[NEXP];
    #pragma unroll
    for (int e = 0; e < NEXP; ++e) s[e] = 0.f;
    #pragma unroll
    for (int i = 0; i < 16; ++i) {
        float xv = xr[lane * 16 + i];
        #pragma unroll
        for (int e = 0; e < NEXP; ++e) s[e] += xv * sw[e][lane * 16 + i];
    }
    #pragma unroll
    for (int e = 0; e < NEXP; ++e) {
        #pragma unroll
        for (int off = 16; off > 0; off >>= 1)
            s[e] += __shfl_xor(s[e], off, 32);
    }
    if (lane == 0) {
        float lg[NEXP], m = -1e30f;
        #pragma unroll
        for (int e = 0; e < NEXP; ++e) { lg[e] = s[e] + sw_b[e]; m = fmaxf(m, lg[e]); }
        float p[NEXP], sum = 0.f;
        #pragma unroll
        for (int e = 0; e < NEXP; ++e) { p[e] = expf(lg[e] - m); sum += p[e]; }
        float inv = 1.f / sum, pm = -1.f; int arg = 0;
        #pragma unroll
        for (int e = 0; e < NEXP; ++e) {
            p[e] *= inv;
            probs[(size_t)t * NEXP + e] = p[e];
            if (p[e] > pm) { pm = p[e]; arg = e; }
        }
        pmax_out[t] = pm;
        routes[t]   = arg;
    }
}

// -------------------- 2) Histogram + offsets (deterministic) --------------------
__global__ __launch_bounds__(256) void hist_kernel(
    const int* __restrict__ routes, int* cnts, int* offs, int* cursor,
    float* out_counts, float* out_zero)
{
    __shared__ int h[NEXP];
    int tid = threadIdx.x;
    if (tid < NEXP) h[tid] = 0;
    __syncthreads();
    for (int i = tid; i < T_TOK; i += 256) atomicAdd(&h[routes[i]], 1);
    __syncthreads();
    if (tid == 0) {
        int off = 0;
        for (int e = 0; e < NEXP; ++e) {
            cnts[e] = h[e]; offs[e] = off; cursor[e] = off; off += h[e];
            out_counts[e] = (float)h[e];
        }
        out_zero[0] = 0.f;
    }
}

// -------------------- 3) Scatter tokens into per-expert lists --------------------
__global__ __launch_bounds__(256) void scatter_kernel(
    const int* __restrict__ routes, int* cursor, int* __restrict__ perm)
{
    int i = blockIdx.x * 256 + threadIdx.x;
    if (i < T_TOK) {
        int e = routes[i];
        int slot = atomicAdd(&cursor[e], 1);
        perm[slot] = i;
    }
}

// -------------------- 4) Deterministic prob-sum reduction --------------------
__global__ __launch_bounds__(256) void probsum_kernel(
    const float* __restrict__ probs, float* __restrict__ out_probsum)
{
    __shared__ float red[256];
    int e = blockIdx.x, tid = threadIdx.x;
    float s = 0.f;
    for (int i = tid; i < T_TOK; i += 256) s += probs[(size_t)i * NEXP + e];
    red[tid] = s; __syncthreads();
    for (int st = 128; st > 0; st >>= 1) {
        if (tid < st) red[tid] += red[tid + st];
        __syncthreads();
    }
    if (tid == 0) out_probsum[e] = red[0];
}

// -------------------- 5) Expert FFN via BF16 WMMA --------------------
__global__ __launch_bounds__(256) void ffn_kernel(
    const float* __restrict__ w1, const float* __restrict__ b1,
    const float* __restrict__ w2, const float* __restrict__ b2,
    const float* __restrict__ x,  const float* __restrict__ pmax,
    const int* __restrict__ perm, const int* __restrict__ cnts,
    const int* __restrict__ offs, float* __restrict__ out)
{
    int e = blockIdx.y;
    int cnt = cnts[e];
    int tile_base = blockIdx.x * M_TILE;
    if (tile_base >= cnt) return;               // block-uniform exit
    int off = offs[e];

    __shared__ __bf16 xs[M_TILE][DMODEL + 8];   // x tile, bf16
    __shared__ __bf16 hs[M_TILE][FF_CHUNK + 8]; // relu(h) chunk, bf16
    __shared__ int   tok[M_TILE];
    __shared__ float pmx[M_TILE];

    int tid  = threadIdx.x;
    int lane = tid & 31, wv = tid >> 5;         // 8 waves

    for (int i = tid; i < M_TILE; i += 256) {
        int idx = tile_base + i;
        int t = perm[off + ((idx < cnt) ? idx : tile_base)];  // clamp tail rows
        tok[i] = t; pmx[i] = pmax[t];
    }
    __syncthreads();

    { // gather + f32->bf16 convert x tile: 4 threads per row
        int row = tid >> 2;
        int cb  = (tid & 3) * 128;
        const float* xr = x + (size_t)tok[row] * DMODEL + cb;
        #pragma unroll 8
        for (int j = 0; j < 128; ++j) xs[row][cb + j] = (__bf16)xr[j];
    }
    __syncthreads();

    const float* w1e = w1 + (size_t)e * DFF * DMODEL;
    const float* w2e = w2 + (size_t)e * DMODEL * DFF;
    const float* b1e = b1 + (size_t)e * DFF;
    const float* b2e = b2 + (size_t)e * DMODEL;

    const int nsel  = lane & 15;
    const int ahalf = (lane < 16) ? 0 : 8;      // A frag: K {0..7,16..23} / {8..15,24..31}
    const int bhalf = (lane < 16) ? 0 : 16;     // B frag: K {0..7,16..23} / {8..15,24..31} (via ahalf)

    v8f yacc[4][4];
    #pragma unroll
    for (int ms = 0; ms < 4; ++ms)
        #pragma unroll
        for (int ns = 0; ns < 4; ++ns) yacc[ms][ns] = (v8f){};

    for (int chunk = 0; chunk < DFF; chunk += FF_CHUNK) {
        // ---- stage 1: h[64 x 128] = x @ w1_chunk^T ; wave wv owns 16 ff cols
        v8f hacc[4];
        #pragma unroll
        for (int ms = 0; ms < 4; ++ms) hacc[ms] = (v8f){};
        int ffb = chunk + wv * 16;
        const float* w1row = w1e + (size_t)(ffb + nsel) * DMODEL;
        for (int ks = 0; ks < 16; ++ks) {       // K = 512 in steps of 32
            v16bf bf;
            #pragma unroll
            for (int i = 0; i < 16; ++i) bf[i] = (__bf16)w1row[ks * 32 + ahalf + (i & 7) + ((i >> 3) << 4)];
            #pragma unroll
            for (int ms = 0; ms < 4; ++ms) {
                const __bf16* ar = xs[ms * 16 + nsel];
                int kb = ks * 32 + ahalf;
                v16bf af;
                #pragma unroll
                for (int i = 0; i < 8; ++i) { af[i] = ar[kb + i]; af[8 + i] = ar[kb + 16 + i]; }
                hacc[ms] = __builtin_amdgcn_wmma_f32_16x16x32_bf16(
                    false, af, false, bf, (short)0, hacc[ms], false, false);
            }
        }
        float b1v = b1e[ffb + nsel];
        #pragma unroll
        for (int ms = 0; ms < 4; ++ms)
            #pragma unroll
            for (int r = 0; r < 8; ++r) {
                float v = hacc[ms][r] + b1v;
                v = v > 0.f ? v : 0.f;          // relu
                hs[ms * 16 + r + ahalf][wv * 16 + nsel] = (__bf16)v;
            }
        __syncthreads();

        // ---- stage 2: y += relu(h) @ w2_chunk^T ; wave wv owns 64 y cols
        int ncb = wv * 64;
        for (int ks = 0; ks < 4; ++ks) {        // K = 128 in steps of 32
            v16bf af[4];
            #pragma unroll
            for (int ms = 0; ms < 4; ++ms) {
                const __bf16* ar = hs[ms * 16 + nsel];
                int kb = ks * 32 + ahalf;
                #pragma unroll
                for (int i = 0; i < 8; ++i) { af[ms][i] = ar[kb + i]; af[ms][8 + i] = ar[kb + 16 + i]; }
            }
            #pragma unroll
            for (int ns = 0; ns < 4; ++ns) {
                const float* w2row = w2e + (size_t)(ncb + ns * 16 + nsel) * DFF + chunk;
                v16bf bf;
                #pragma unroll
                for (int i = 0; i < 16; ++i) bf[i] = (__bf16)w2row[ks * 32 + ahalf + (i & 7) + ((i >> 3) << 4)];
                #pragma unroll
                for (int ms = 0; ms < 4; ++ms)
                    yacc[ms][ns] = __builtin_amdgcn_wmma_f32_16x16x32_bf16(
                        false, af[ms], false, bf, (short)0, yacc[ms][ns], false, false);
            }
        }
        __syncthreads();                        // before next chunk rewrites hs
    }

    // ---- epilogue: (y + b2) * p_max, masked store
    int ncb = wv * 64;
    #pragma unroll
    for (int ns = 0; ns < 4; ++ns) {
        float b2v = b2e[ncb + ns * 16 + nsel];
        #pragma unroll
        for (int ms = 0; ms < 4; ++ms)
            #pragma unroll
            for (int r = 0; r < 8; ++r) {
                int row = ms * 16 + r + ahalf;
                if (tile_base + row < cnt) {
                    out[(size_t)tok[row] * DMODEL + ncb + ns * 16 + nsel] =
                        (yacc[ms][ns][r] + b2v) * pmx[row];
                }
            }
    }
}

// -------------------- launch --------------------
extern "C" void kernel_launch(void* const* d_in, const int* in_sizes, int n_in,
                              void* d_out, int out_size, void* d_ws, size_t ws_size,
                              hipStream_t stream) {
    const float* x    = (const float*)d_in[0];
    const float* sw_w = (const float*)d_in[1];
    const float* sw_b = (const float*)d_in[2];
    const float* w1   = (const float*)d_in[3];
    const float* b1   = (const float*)d_in[4];
    const float* w2   = (const float*)d_in[5];
    const float* b2   = (const float*)d_in[6];

    float* out         = (float*)d_out;            // final [16384,512]
    float* out_counts  = out + (size_t)T_TOK * DMODEL;  // [8]
    float* out_probsum = out_counts + NEXP;             // [8]
    float* out_zero    = out_probsum + NEXP;            // [1]
    float* out_pmax    = out_zero + 1;                  // [16384]

    int*   routes = (int*)d_ws;          // T
    int*   perm   = routes + T_TOK;      // T
    int*   cnts   = perm + T_TOK;        // 8
    int*   offs   = cnts + NEXP;         // 8
    int*   cursor = offs + NEXP;         // 8
    float* probs  = (float*)(cursor + NEXP); // T*8

    router_kernel<<<T_TOK / 8, 256, 0, stream>>>(x, sw_w, sw_b, out_pmax, probs, routes);
    hist_kernel<<<1, 256, 0, stream>>>(routes, cnts, offs, cursor, out_counts, out_zero);
    scatter_kernel<<<T_TOK / 256, 256, 0, stream>>>(routes, cursor, perm);
    probsum_kernel<<<NEXP, 256, 0, stream>>>(probs, out_probsum);

    dim3 g(T_TOK / M_TILE, NEXP);
    ffn_kernel<<<g, 256, 0, stream>>>(w1, b1, w2, b2, x, out_pmax, perm, cnts, offs, out);
}
